// TinyOnnModel_76089640616234
// MI455X (gfx1250) — matrix operand order repaired and checked by reading, not compiled, not run on hardware
//
#include <hip/hip_runtime.h>
#include <hip/hip_bf16.h>
#include <math.h>

typedef __bf16 bf16_t;
typedef __attribute__((ext_vector_type(16))) __bf16 v16bf;
typedef __attribute__((ext_vector_type(8)))  float  v8f;

#define BB 4
#define TT 1024
#define CC 1024
#define II 2048
#define EE 8
#define LL 4
#define NN (BB*TT)   /* 4096 tokens */

// ---------------------------------------------------------------------------
// Embedding: x[t,c] = tok_embed[ids[t],c] + pos_embed[t % T, c]
// ---------------------------------------------------------------------------
__global__ void embed_kernel(const int* __restrict__ ids,
                             const float* __restrict__ tok,
                             const float* __restrict__ pos,
                             float* __restrict__ x) {
  int t = blockIdx.x;
  int row = ids[t];
  int pt = t & (TT - 1);
  const float* tr = tok + (size_t)row * CC;
  const float* pr = pos + (size_t)pt * CC;
  float* xr = x + (size_t)t * CC;
  for (int c = threadIdx.x; c < CC; c += blockDim.x)
    xr[c] = tr[c] + pr[c];
}

// ---------------------------------------------------------------------------
// LayerNorm per token; optionally also emit bf16 copy for WMMA A-matrix.
// ---------------------------------------------------------------------------
template<bool WRITE_BF>
__global__ void ln_kernel(const float* __restrict__ x, const float* __restrict__ w,
                          const float* __restrict__ b, float* __restrict__ nf,
                          bf16_t* __restrict__ nbf) {
  __shared__ float red[256];
  int t = blockIdx.x, tid = threadIdx.x;
  const float* xr = x + (size_t)t * CC;

  float s = 0.f;
  for (int c = tid; c < CC; c += 256) s += xr[c];
  red[tid] = s; __syncthreads();
  for (int off = 128; off; off >>= 1) {
    if (tid < off) red[tid] += red[tid + off];
    __syncthreads();
  }
  float mu = red[0] * (1.0f / CC);
  __syncthreads();

  float v = 0.f;
  for (int c = tid; c < CC; c += 256) { float d = xr[c] - mu; v += d * d; }
  red[tid] = v; __syncthreads();
  for (int off = 128; off; off >>= 1) {
    if (tid < off) red[tid] += red[tid + off];
    __syncthreads();
  }
  float inv = rsqrtf(red[0] * (1.0f / CC) + 1e-5f);

  for (int c = tid; c < CC; c += 256) {
    float val = (xr[c] - mu) * inv * w[c] + b[c];
    nf[(size_t)t * CC + c] = val;
    if (WRITE_BF) nbf[(size_t)t * CC + c] = (bf16_t)val;
  }
}

// ---------------------------------------------------------------------------
// Per-layer sim column norms: simn[e] = max(||sim[:,e]||, 1e-12)
// ---------------------------------------------------------------------------
__global__ void simnorm_kernel(const float* __restrict__ sim, float* __restrict__ simn) {
  int tid = threadIdx.x;
  int e = tid >> 5, lane = tid & 31;
  float s = 0.f;
  for (int c = lane; c < CC; c += 32) { float v = sim[(size_t)c * EE + e]; s += v * v; }
  for (int off = 16; off; off >>= 1) s += __shfl_xor(s, off, 32);
  if (lane == 0) simn[e] = fmaxf(sqrtf(s), 1e-12f);
}

// ---------------------------------------------------------------------------
// Gating: cosine logits vs sim columns, ReLU mask, top-2 fallback, softmax.
// ---------------------------------------------------------------------------
__global__ void gate_kernel(const float* __restrict__ nf, const float* __restrict__ sim,
                            const float* __restrict__ gates, const float* __restrict__ simn,
                            float* __restrict__ rw) {
  __shared__ float sh[9][256];
  int t = blockIdx.x, tid = threadIdx.x;
  const float* xr = nf + (size_t)t * CC;

  float dot[EE]; float nrm = 0.f;
  for (int e = 0; e < EE; e++) dot[e] = 0.f;
  for (int c = tid; c < CC; c += 256) {
    float xv = xr[c];
    nrm += xv * xv;
    for (int e = 0; e < EE; e++) dot[e] += xv * sim[(size_t)c * EE + e];
  }
  for (int e = 0; e < EE; e++) sh[e][tid] = dot[e];
  sh[8][tid] = nrm;
  __syncthreads();
  for (int off = 128; off; off >>= 1) {
    if (tid < off) {
      for (int q = 0; q < 9; q++) sh[q][tid] += sh[q][tid + off];
    }
    __syncthreads();
  }

  if (tid == 0) {
    float xn = fmaxf(sqrtf(sh[8][0]), 1e-12f);
    float logits[EE], gated[EE];
    bool mask[EE];
    bool any = false;
    for (int e = 0; e < EE; e++) {
      float lg = sh[e][0] / (xn * simn[e]) - 1.0f / (1.0f + expf(-gates[e]));
      logits[e] = lg;
      gated[e] = fmaxf(lg, 0.f);
      mask[e] = gated[e] > 0.f;
      if (mask[e]) any = true;
    }
    if (!any) {
      int i0 = 0;
      for (int e = 1; e < EE; e++) if (logits[e] > logits[i0]) i0 = e;
      int i1 = -1;
      for (int e = 0; e < EE; e++) {
        if (e == i0) continue;
        if (i1 < 0 || logits[e] > logits[i1]) i1 = e;
      }
      mask[i0] = true; mask[i1] = true;
    }
    float mx = -1e30f;
    for (int e = 0; e < EE; e++) if (mask[e] && gated[e] > mx) mx = gated[e];
    float ex[EE]; float ssum = 0.f;
    for (int e = 0; e < EE; e++) { ex[e] = mask[e] ? expf(gated[e] - mx) : 0.f; ssum += ex[e]; }
    for (int e = 0; e < EE; e++) rw[(size_t)t * EE + e] = ex[e] / ssum;
  }
}

// ---------------------------------------------------------------------------
// fp32 -> bf16 convert + transpose:  src [E][R][Cc]  ->  dst [E][Cc][R]
// Weights are read once per layer (HBM-bound step); transposed so the GEMM's
// B matrix is K-contiguous (enables pure B128 async tile copies, no scalar
// LDS scatter). Tiled 32x32 via padded LDS for coalescing.
// ---------------------------------------------------------------------------
__global__ __launch_bounds__(256) void cvt_transpose_kernel(
    const float* __restrict__ src, bf16_t* __restrict__ dst, int R, int Cc) {
  __shared__ float tile[32][33];
  int e = blockIdx.z;
  int r0 = blockIdx.y * 32;
  int c0 = blockIdx.x * 32;
  const float* s = src + (size_t)e * R * Cc;
  bf16_t* d = dst + (size_t)e * Cc * R;
  int tid = threadIdx.x;
  int lr = tid >> 3;          // 0..31
  int lc4 = (tid & 7) * 4;    // 0,4,...,28
#pragma unroll
  for (int j = 0; j < 4; j++)
    tile[lr][lc4 + j] = s[(size_t)(r0 + lr) * Cc + c0 + lc4 + j];
  __syncthreads();
#pragma unroll
  for (int j = 0; j < 4; j++)
    d[(size_t)(c0 + lr) * R + r0 + lc4 + j] = (bf16_t)tile[lc4 + j][lr];
}

// ---------------------------------------------------------------------------
// Soft-MoE GEMM:  out[t,n] = sum_e rw[t,e] * (A[t,:] @ W_e[n,:])   (W N-major)
//   GELU=true : gelu -> bf16 store (h for GEMM2);  GELU=false: outAdd += val.
// WG = 256 thr = 8 waves (4M x 2N); WG tile 64x128, wave tile 16x64,
// K-step 32 via v_wmma_f32_16x16x32_bf16.
// Double-buffered LDS tiles filled with GLOBAL_LOAD_ASYNC_TO_LDS_B128;
// per K-step: s_wait_asynccnt 0 -> barrier -> issue next slab -> 4x WMMA.
// ---------------------------------------------------------------------------
template<int KT, int NT, bool GELU>
__global__ __launch_bounds__(256) void moe_gemm_kernel(
    const bf16_t* __restrict__ A, const bf16_t* __restrict__ W,
    const float* __restrict__ rw, bf16_t* __restrict__ outBf,
    float* __restrict__ outAdd) {

  __shared__ __align__(16) bf16_t Asm[2][64 * 32];
  __shared__ __align__(16) bf16_t Bsm[2][128 * 32];   // [col][k]
  __shared__ float rwsm[64 * EE];

  int tid = threadIdx.x;
  int wave = tid >> 5, lane = tid & 31;
  int m0 = blockIdx.y * 64;
  int n0 = blockIdx.x * 128;
  int waveM = wave >> 1, waveN = wave & 1;
  int hi = lane >> 4;
  int l16 = lane & 15;

  for (int i = tid; i < 64 * EE; i += 256)
    rwsm[i] = rw[(size_t)(m0 + (i >> 3)) * EE + (i & 7)];

  int arow = tid >> 2;          // 0..63
  int akc  = (tid & 3) * 8;     // 0,8,16,24

  v8f accSum[4];
#pragma unroll
  for (int s = 0; s < 4; s++)
#pragma unroll
    for (int j = 0; j < 8; j++) accSum[s][j] = 0.f;

  for (int e = 0; e < EE; e++) {
    const bf16_t* We = W + (size_t)e * (size_t)KT * NT;   // [n][k], n stride KT

    // async-issue one 64x32 A slab + one 128x32 B slab into LDS buffer p
    auto issueTile = [&](int k0, int p) {
      // A: one 16B chunk per thread
      {
        unsigned lds = (unsigned)(uintptr_t)&Asm[p][arow * 32 + akc];
        unsigned long long g =
            (unsigned long long)&A[(size_t)(m0 + arow) * KT + k0 + akc];
        asm volatile("global_load_async_to_lds_b128 %0, %1, off"
                     :: "v"(lds), "v"(g) : "memory");
      }
      // B: two 16B chunks per thread (K-contiguous thanks to transposed W)
#pragma unroll
      for (int rep = 0; rep < 2; rep++) {
        int idx = tid + rep * 256;     // 0..511
        int col = idx >> 2;            // 0..127
        int kc  = (idx & 3) * 8;       // 0,8,16,24
        unsigned lds = (unsigned)(uintptr_t)&Bsm[p][col * 32 + kc];
        unsigned long long g =
            (unsigned long long)&We[(size_t)(n0 + col) * KT + k0 + kc];
        asm volatile("global_load_async_to_lds_b128 %0, %1, off"
                     :: "v"(lds), "v"(g) : "memory");
      }
      // speculative L2 prefetch of the slab after next
      __builtin_prefetch(&A[(size_t)(m0 + arow) * KT + k0 + 64 + akc], 0, 1);
    };

    v8f acc[4];
#pragma unroll
    for (int s = 0; s < 4; s++)
#pragma unroll
      for (int j = 0; j < 8; j++) acc[s][j] = 0.f;

    constexpr int NIT = KT / 32;
    issueTile(0, 0);
    for (int it = 0; it < NIT; it++) {
      int p = it & 1;
      // my slab-`it` transfers complete
      asm volatile("s_wait_asynccnt 0x0" ::: "memory");
      // everyone's slab-`it` in LDS; everyone done reading buffer p^1
      __syncthreads();
      if (it + 1 < NIT) issueTile((it + 1) * 32, p ^ 1);

      // A fragment (16x32 bf16, ISA 7.12.2): lanes<16 K{0..7,16..23}, else +8
      union { uint4 u[2]; v16bf v; } af;
      {
        const uint4* ap = (const uint4*)&Asm[p][(waveM * 16 + l16) * 32 + hi * 8];
        af.u[0] = ap[0];
        af.u[1] = ap[2];
      }
#pragma unroll
      for (int s = 0; s < 4; s++) {
        // B fragment (32x16): N=l16, lanes<16 K{0..7,16..23}, else +8
        union { uint4 u[2]; v16bf v; } bfg;
        const uint4* bp =
            (const uint4*)&Bsm[p][(waveN * 64 + s * 16 + l16) * 32 + hi * 8];
        bfg.u[0] = bp[0];
        bfg.u[1] = bp[2];
        acc[s] = __builtin_amdgcn_wmma_f32_16x16x32_bf16(
            false, af.v, false, bfg.v, (short)0, acc[s], false, false);
      }
    }

    // fold expert partial into running sum scaled by rw[t,e]
#pragma unroll
    for (int s = 0; s < 4; s++)
#pragma unroll
      for (int j = 0; j < 8; j++) {
        int mrow = waveM * 16 + j + hi * 8;
        accSum[s][j] += rwsm[mrow * EE + e] * acc[s][j];
      }
  }

  // epilogue
#pragma unroll
  for (int s = 0; s < 4; s++)
#pragma unroll
    for (int j = 0; j < 8; j++) {
      int gm = m0 + waveM * 16 + j + hi * 8;
      int gn = n0 + waveN * 64 + s * 16 + l16;
      float vv = accSum[s][j];
      if (GELU) {
        float g = 0.5f * vv * (1.0f + erff(vv * 0.70710678118654752f));
        outBf[(size_t)gm * NT + gn] = (bf16_t)g;
      } else {
        outAdd[(size_t)gm * NT + gn] += vv;
      }
    }
}

// ---------------------------------------------------------------------------
extern "C" void kernel_launch(void* const* d_in, const int* in_sizes, int n_in,
                              void* d_out, int out_size, void* d_ws, size_t ws_size,
                              hipStream_t stream) {
  (void)in_sizes; (void)n_in; (void)out_size; (void)ws_size;
  const int*   ids   = (const int*)d_in[0];
  const float* tok   = (const float*)d_in[1];
  const float* pos   = (const float*)d_in[2];
  const float* lnw   = (const float*)d_in[3];
  const float* lnb   = (const float*)d_in[4];
  const float* sim   = (const float*)d_in[5];
  const float* gates = (const float*)d_in[6];
  const float* w1    = (const float*)d_in[7];
  const float* w2    = (const float*)d_in[8];
  const float* flnw  = (const float*)d_in[9];
  const float* flnb  = (const float*)d_in[10];

  char* ws = (char*)d_ws;
  size_t off = 0;
  auto alloc = [&](size_t bytes) -> void* {
    void* p = ws + off;
    off += (bytes + 255) & ~(size_t)255;
    return p;
  };
  float*  xbuf = (float*) alloc((size_t)NN * CC * 4);
  float*  nf   = (float*) alloc((size_t)NN * CC * 4);
  bf16_t* nbf  = (bf16_t*)alloc((size_t)NN * CC * 2);
  bf16_t* hbf  = (bf16_t*)alloc((size_t)NN * II * 2);
  bf16_t* w1t  = (bf16_t*)alloc((size_t)EE * CC * II * 2);  // [e][i][c]
  bf16_t* w2t  = (bf16_t*)alloc((size_t)EE * II * CC * 2);  // [e][c][i]
  float*  rwb  = (float*) alloc((size_t)NN * EE * 4);
  float*  simn = (float*) alloc(256);

  embed_kernel<<<NN, 256, 0, stream>>>(ids, tok, pos, xbuf);

  const size_t wcnt = (size_t)EE * CC * II;
  for (int l = 0; l < LL; l++) {
    ln_kernel<true><<<NN, 256, 0, stream>>>(
        xbuf, lnw + (size_t)l * CC, lnb + (size_t)l * CC, nf, nbf);
    simnorm_kernel<<<1, 256, 0, stream>>>(sim + (size_t)l * CC * EE, simn);
    gate_kernel<<<NN, 256, 0, stream>>>(
        nf, sim + (size_t)l * CC * EE, gates + (size_t)l * EE, simn, rwb);
    // w1 [e][c][i] -> w1t [e][i][c];  w2 [e][i][c] -> w2t [e][c][i]
    cvt_transpose_kernel<<<dim3(II / 32, CC / 32, EE), 256, 0, stream>>>(
        w1 + (size_t)l * wcnt, w1t, CC, II);
    cvt_transpose_kernel<<<dim3(CC / 32, II / 32, EE), 256, 0, stream>>>(
        w2 + (size_t)l * wcnt, w2t, II, CC);
    // GEMM1: M=4096, K=C=1024, N=I=2048, gelu -> h (bf16)
    moe_gemm_kernel<CC, II, true><<<dim3(II / 128, NN / 64), 256, 0, stream>>>(
        nbf, w1t, rwb, hbf, nullptr);
    // GEMM2: M=4096, K=I=2048, N=C=1024, residual add into x
    moe_gemm_kernel<II, CC, false><<<dim3(CC / 128, NN / 64), 256, 0, stream>>>(
        hbf, w2t, rwb, nullptr, xbuf);
  }

  ln_kernel<false><<<NN, 256, 0, stream>>>(xbuf, flnw, flnb, (float*)d_out, nullptr);
}
